// UniAffineAttention_34763465294573
// MI455X (gfx1250) — matrix operand order repaired and checked by reading, not compiled, not run on hardware
//
#include <hip/hip_runtime.h>
#include <hip/hip_bf16.h>

// ---------------------------------------------------------------------------
// UniAffineAttention for MI455X (gfx1250): bf16 WMMA everywhere, f32 accum.
// B=2, S=2048, HID=2048, H=16, KVH=4, D=128, N_QKV=3072, M=B*S=4096.
// GEMM wave tile = 32(M) x 128(N) with __launch_bounds__(256,1) so the
// 128 accumulator VGPRs stay register-resident (no scratch spills).
// ---------------------------------------------------------------------------

typedef __bf16 bf16_t;
typedef __attribute__((ext_vector_type(8)))  bf16_t bf16x8;
typedef __attribute__((ext_vector_type(16))) bf16_t bf16x16;
typedef __attribute__((ext_vector_type(8)))  float  f32x8;

#define WMMA_BF16(A, B, C) \
  __builtin_amdgcn_wmma_f32_16x16x32_bf16(false, (A), false, (B), (short)0, (C), false, false)

#define SCALE_F 2.1579187e-05f   // 1/(sqrt(128)*4096)

// --- A-fragment (16x32 bf16, M x K) from row-major src (row=M, contiguous K).
// ISA layout: lanes 0-15: row=lane, K chunks {0..7, 16..23};
//             lanes 16-31: row=lane-16, K chunks {8..15, 24..31}.
__device__ __forceinline__ bf16x16 load_a16(const bf16_t* base, int ld, int lane) {
  const int row  = lane & 15;
  const int koff = (lane >> 4) << 3;   // 0 or 8
  const bf16_t* p = base + row * ld + koff;
  bf16x8 lo = *(const bf16x8*)(p);
  bf16x8 hi = *(const bf16x8*)(p + 16);
  bf16x16 r;
#pragma unroll
  for (int i = 0; i < 8; ++i) { r[i] = lo[i]; r[i + 8] = hi[i]; }
  return r;
}

// --- B-fragment (32x16 bf16, K x N) from "n-major" src (row=N, contiguous K).
// ISA layout: lanes 0-15: col=lane, K chunks {0..7, 16..23}; lanes 16-31: K chunks {8..15, 24..31}.
__device__ __forceinline__ bf16x16 load_b16(const bf16_t* base_nmajor, int ld, int lane) {
  const int n    = lane & 15;
  const int koff = (lane >> 4) << 3;   // 0 or 8
  const bf16_t* p = base_nmajor + n * ld + koff;
  bf16x8 lo = *(const bf16x8*)(p);
  bf16x8 hi = *(const bf16x8*)(p + 16);
  bf16x16 r;
#pragma unroll
  for (int i = 0; i < 8; ++i) { r[i] = lo[i]; r[i + 8] = hi[i]; }
  return r;
}

// ---------------------------------------------------------------------------
// f32 -> bf16 bulk convert (4 elems/thread)
__global__ void k_cvt_bf16(const float* __restrict__ src, bf16_t* __restrict__ dst) {
  int i = (blockIdx.x * 256 + threadIdx.x) * 4;
  float4 v = *(const float4*)(src + i);
  dst[i + 0] = (bf16_t)v.x;
  dst[i + 1] = (bf16_t)v.y;
  dst[i + 2] = (bf16_t)v.z;
  dst[i + 3] = (bf16_t)v.w;
}

// Wo [d=2048][o=2048] f32  ->  WoT [o][d] bf16 (so GEMM B-frags read contiguous d)
__global__ void k_wo_T(const float* __restrict__ wo, bf16_t* __restrict__ woT) {
  int idx = blockIdx.x * 256 + threadIdx.x;      // idx = d*2048 + o (coalesced read)
  int d = idx >> 11, o = idx & 2047;
  woT[o * 2048 + d] = (bf16_t)wo[idx];
}

// ---------------------------------------------------------------------------
// QKV GEMM: qkv = x @ Wqkv^T   (M=4096, N=3072, K=2048), fused RoPE + layout.
// Wave tile: 32(M) x 128(N); 128 N-cols = exactly one head slot. 8 waves/block.
__global__ void __launch_bounds__(256, 1)
k_gemm_qkv(const bf16_t* __restrict__ xb,
           const bf16_t* __restrict__ wb,
           bf16_t* __restrict__ qb,   // [B,16,S,128]
           bf16_t* __restrict__ kbuf, // [B, 4,S,128]
           bf16_t* __restrict__ vtb)  // [B, 4,128,S] (transposed)
{
  const int lane = threadIdx.x & 31;
  const int wv   = threadIdx.x >> 5;
  const int wid  = blockIdx.x * 8 + wv;
  const int mt   = wid & 127;   // 128 M-tiles of 32 rows
  const int nt   = wid >> 7;    // 24 head slots of 128 cols

  const bf16_t* abase = xb + (size_t)mt * 32 * 2048;
  const bf16_t* bbase = wb + (size_t)nt * 128 * 2048;

  f32x8 acc[2][8] = {};
  for (int k0 = 0; k0 < 2048; k0 += 32) {
    bf16x16 a0 = load_a16(abase + k0, 2048, lane);
    bf16x16 a1 = load_a16(abase + 16 * 2048 + k0, 2048, lane);
#pragma unroll
    for (int j = 0; j < 8; ++j) {
      bf16x16 b = load_b16(bbase + j * 16 * 2048 + k0, 2048, lane);
      acc[0][j] = WMMA_BF16(a0, b, acc[0][j]);
      acc[1][j] = WMMA_BF16(a1, b, acc[1][j]);
    }
  }

  // Epilogue. C layout: lane<16 -> (M=g, N=lane); lane>=16 -> (M=g+8, N=lane-16).
  const int n    = lane & 15;
  const int half = lane >> 4;

  if (nt < 20) {
    // q (nt<16) or k (16<=nt<20): apply RoPE in-register.
    bf16_t* dstbase;
    int heads_stride;
    if (nt < 16) { dstbase = qb;   heads_stride = 16; }
    else         { dstbase = kbuf; heads_stride = 4;  }
    const int h = (nt < 16) ? nt : (nt - 16);
#pragma unroll
    for (int mi = 0; mi < 2; ++mi) {
      const int row0 = mt * 32 + mi * 16;
#pragma unroll
      for (int j = 0; j < 4; ++j) {
        const int dlo = j * 16 + n;                       // 0..63
        const float invf = __powf(10000.0f, -(float)dlo * (1.0f / 64.0f));
#pragma unroll
        for (int g = 0; g < 8; ++g) {
          const int r = row0 + g + (half << 3);
          const int bb = r >> 11, s = r & 2047;
          const float ang = (float)s * invf;
          const float c = __cosf(ang), si = __sinf(ang);
          const float x1 = acc[mi][j][g], x2 = acc[mi][j + 4][g];
          bf16_t* dst = dstbase + ((size_t)((bb * heads_stride + h) * 2048 + s)) * 128;
          dst[dlo]      = (bf16_t)(x1 * c - x2 * si);
          dst[dlo + 64] = (bf16_t)(x2 * c + x1 * si);
        }
      }
    }
  } else {
    // v: store transposed [B,KVH,D,S], no RoPE.
    const int hv = nt - 20;
#pragma unroll
    for (int mi = 0; mi < 2; ++mi) {
      const int row0 = mt * 32 + mi * 16;
#pragma unroll
      for (int j = 0; j < 8; ++j) {
        const int d = j * 16 + n;
#pragma unroll
        for (int g = 0; g < 8; ++g) {
          const int r = row0 + g + (half << 3);
          const int bb = r >> 11, s = r & 2047;
          vtb[((size_t)(bb * 4 + hv) * 128 + d) * 2048 + s] = (bf16_t)acc[mi][j][g];
        }
      }
    }
  }
}

// ---------------------------------------------------------------------------
// Attention: per wave 16 queries; stream 32-key blocks with causal cutoff.
// scores -> affine clamp -> bf16 via per-wave LDS tile -> WMMA with V^T.
__global__ void k_attn(const bf16_t* __restrict__ qb,
                       const bf16_t* __restrict__ kbuf,
                       const bf16_t* __restrict__ vtb,
                       const float* __restrict__ act_a,
                       const float* __restrict__ act_b,
                       const float* __restrict__ act_g,
                       bf16_t* __restrict__ attnb)    // [B*S, 2048]
{
  __shared__ alignas(16) bf16_t wlds[8][16][32];

  const int lane = threadIdx.x & 31;
  const int wv   = threadIdx.x >> 5;
  const int qt   = blockIdx.x & 15;         // 16 query tiles of 128
  const int h    = (blockIdx.x >> 4) & 15;  // 16 heads
  const int b    = blockIdx.x >> 8;         // batch
  const int hv   = h >> 2;                  // GQA: 4 q-heads per kv-head
  const int s0   = qt * 128 + wv * 16;      // this wave's first query row

  const bf16_t* qbase = qb   + ((size_t)(b * 16 + h) * 2048 + s0) * 128;
  const bf16_t* kbase = kbuf + ((size_t)(b * 4 + hv) * 2048) * 128;
  const bf16_t* vbase = vtb  + ((size_t)(b * 4 + hv) * 128) * 2048;

  const float aH = act_a[h], bH = act_b[h], gH = act_g[h];

  bf16x16 qa[4];
#pragma unroll
  for (int kk = 0; kk < 4; ++kk) qa[kk] = load_a16(qbase + kk * 32, 128, lane);

  f32x8 o[8] = {};
  const int n    = lane & 15;
  const int half = lane >> 4;

  for (int t0 = 0; t0 < s0 + 16; t0 += 32) {
    // scores: 16 queries x 32 keys (two 16x16 C fragments), K = D = 128
    f32x8 sc0 = {}, sc1 = {};
#pragma unroll
    for (int kk = 0; kk < 4; ++kk) {
      bf16x16 kb0 = load_b16(kbase + (size_t)t0 * 128 + kk * 32, 128, lane);
      bf16x16 kb1 = load_b16(kbase + (size_t)(t0 + 16) * 128 + kk * 32, 128, lane);
      sc0 = WMMA_BF16(qa[kk], kb0, sc0);
      sc1 = WMMA_BF16(qa[kk], kb1, sc1);
    }
    // epilogue: scale + causal mask + gamma*clamp(relu(a*s+b),0,1) -> LDS bf16
#pragma unroll
    for (int g = 0; g < 8; ++g) {
      const int m  = s0 + g + (half << 3);
      const int ml = g + (half << 3);
      const int t  = t0 + n;
      float v0 = sc0[g] * SCALE_F + ((t <= m) ? 0.0f : -1e9f);
      wlds[wv][ml][n]      = (bf16_t)(fminf(fmaxf(aH * v0 + bH, 0.0f), 1.0f) * gH);
      float v1 = sc1[g] * SCALE_F + ((t + 16 <= m) ? 0.0f : -1e9f);
      wlds[wv][ml][n + 16] = (bf16_t)(fminf(fmaxf(aH * v1 + bH, 0.0f), 1.0f) * gH);
    }
    // out += w(16x32) @ v(32x128): A from LDS, B from transposed V
    bf16x16 wa = load_a16(&wlds[wv][0][0], 32, lane);
#pragma unroll
    for (int j = 0; j < 8; ++j) {
      bf16x16 vb = load_b16(vbase + (size_t)(j * 16) * 2048 + t0, 2048, lane);
      o[j] = WMMA_BF16(wa, vb, o[j]);
    }
  }

  // store out tile -> attn [b*S+s][h*128+d] bf16
#pragma unroll
  for (int j = 0; j < 8; ++j)
#pragma unroll
    for (int g = 0; g < 8; ++g) {
      const int s = s0 + g + (half << 3);
      attnb[(size_t)(b * 2048 + s) * 2048 + h * 128 + j * 16 + n] = (bf16_t)o[j][g];
    }
}

// ---------------------------------------------------------------------------
// Output GEMM: out = attn @ Wo   (M=4096, N=2048, K=2048), f32 output.
// Wave tile: 32(M) x 128(N).
__global__ void __launch_bounds__(256, 1)
k_gemm_out(const bf16_t* __restrict__ attnb,
           const bf16_t* __restrict__ woTb,
           float* __restrict__ out)
{
  const int lane = threadIdx.x & 31;
  const int wv   = threadIdx.x >> 5;
  const int wid  = blockIdx.x * 8 + wv;
  const int nt   = wid & 15;    // 16 N-tiles of 128
  const int mt   = wid >> 4;    // 128 M-tiles of 32

  const bf16_t* abase = attnb + (size_t)mt * 32 * 2048;
  const bf16_t* bbase = woTb + (size_t)nt * 128 * 2048;

  f32x8 acc[2][8] = {};
  for (int k0 = 0; k0 < 2048; k0 += 32) {
    bf16x16 a0 = load_a16(abase + k0, 2048, lane);
    bf16x16 a1 = load_a16(abase + 16 * 2048 + k0, 2048, lane);
#pragma unroll
    for (int j = 0; j < 8; ++j) {
      bf16x16 b = load_b16(bbase + j * 16 * 2048 + k0, 2048, lane);
      acc[0][j] = WMMA_BF16(a0, b, acc[0][j]);
      acc[1][j] = WMMA_BF16(a1, b, acc[1][j]);
    }
  }
  const int n = lane & 15, half = lane >> 4;
#pragma unroll
  for (int mi = 0; mi < 2; ++mi)
#pragma unroll
    for (int j = 0; j < 8; ++j)
#pragma unroll
      for (int g = 0; g < 8; ++g)
        out[(size_t)(mt * 32 + mi * 16 + g + (half << 3)) * 2048 + nt * 128 + j * 16 + n]
            = acc[mi][j][g];
}

// ---------------------------------------------------------------------------
extern "C" void kernel_launch(void* const* d_in, const int* in_sizes, int n_in,
                              void* d_out, int out_size, void* d_ws, size_t ws_size,
                              hipStream_t stream) {
  (void)in_sizes; (void)n_in; (void)out_size; (void)ws_size;
  const float* x     = (const float*)d_in[0];   // [2,2048,2048]
  const float* Wqkv  = (const float*)d_in[1];   // [3072,2048]
  const float* Wo    = (const float*)d_in[2];   // [2048,2048] (d,o)
  const float* act_a = (const float*)d_in[3];   // [16]
  const float* act_b = (const float*)d_in[4];   // [16]
  const float* gamma = (const float*)d_in[5];   // [16]
  // d_in[6] causal_mask: computed analytically on device, not read.
  float* out = (float*)d_out;

  char* w = (char*)d_ws;
  size_t off = 0;
  bf16_t* xb    = (bf16_t*)(w + off); off += (size_t)4096 * 2048 * 2;         // 16 MB
  bf16_t* wqkvb = (bf16_t*)(w + off); off += (size_t)3072 * 2048 * 2;         // 12 MB
  bf16_t* woTb  = (bf16_t*)(w + off); off += (size_t)2048 * 2048 * 2;         //  8 MB
  bf16_t* qb    = (bf16_t*)(w + off); off += (size_t)2 * 16 * 2048 * 128 * 2; // 16 MB
  bf16_t* kbuf  = (bf16_t*)(w + off); off += (size_t)2 * 4 * 2048 * 128 * 2;  //  4 MB
  bf16_t* vtb   = (bf16_t*)(w + off); off += (size_t)2 * 4 * 2048 * 128 * 2;  //  4 MB
  bf16_t* attnb = (bf16_t*)(w + off); off += (size_t)4096 * 2048 * 2;         // 16 MB

  k_cvt_bf16<<<8192, 256, 0, stream>>>(x, xb);        // 8M elems / 4
  k_cvt_bf16<<<6144, 256, 0, stream>>>(Wqkv, wqkvb);  // 6M elems / 4
  k_wo_T    <<<16384, 256, 0, stream>>>(Wo, woTb);    // 4M elems
  k_gemm_qkv<<<384, 256, 0, stream>>>(xb, wqkvb, qb, kbuf, vtb);   // 3072 waves
  k_attn    <<<512, 256, 0, stream>>>(qb, kbuf, vtb, act_a, act_b, gamma, attnb);
  k_gemm_out<<<256, 256, 0, stream>>>(attnb, woTb, out);           // 2048 waves
}
